// MSAttention_88210038326186
// MI455X (gfx1250) — matrix operand order repaired and checked by reading, not compiled, not run on hardware
//
#include <hip/hip_runtime.h>
#include <hip/hip_bf16.h>
#include <stdint.h>

typedef __attribute__((ext_vector_type(16))) __bf16 v16bf;
typedef __attribute__((ext_vector_type(8)))  float  v8f;

#define DIMC  512
#define LSZ   32
#define NTOK  1024
#define NH    4
#define CCH   128
#define DH    32
#define BB    8
#define EPSF  1e-5f
#define ATT_SCALE 0.17677669529663687f   /* 32^-0.5 */

union ABf  { v16bf v; uint32_t u[8]; };
union AccF { v8f  v; float    f[8]; };

__device__ __forceinline__ uint16_t f2bf(float f) {
  uint32_t u = __float_as_uint(f);
  u += 0x7FFFu + ((u >> 16) & 1u);          // round-to-nearest-even
  return (uint16_t)(u >> 16);
}
// A-matrix (16-bit 16x32) k index for VGPR v, this lane: contiguous k = 16*(lane/16) + 2v (matches B).
__device__ __forceinline__ int a_k0(int v, int lane) {
  int kb = (lane < 16) ? 0 : 16;
  return kb + 2 * v;
}
__device__ __forceinline__ v8f v8f_zero() {
  v8f z;
#pragma unroll
  for (int i = 0; i < 8; ++i) z[i] = 0.f;
  return z;
}
__device__ __forceinline__ float hmax16(float x) {
#pragma unroll
  for (int m = 8; m >= 1; m >>= 1) x = fmaxf(x, __shfl_xor(x, m, 32));
  return x;
}
__device__ __forceinline__ float hsum16(float x) {
#pragma unroll
  for (int m = 8; m >= 1; m >>= 1) x += __shfl_xor(x, m, 32);
  return x;
}

// ---------------- Kernel 0: elementwise fp32 -> bf16 ----------------
__global__ void cvt_bf16(const float* __restrict__ in, uint16_t* __restrict__ out, int n) {
  int i = blockIdx.x * blockDim.x + threadIdx.x;
  if (i < n) out[i] = f2bf(in[i]);
}

// ---------------- Kernel 1: ident = BN1(W_in @ X^T), stored [b][o][n] --------------
// One wave: 16 (o) x 64 (n) slab = 4 WMMA tiles sharing the A (weight) tile.
__global__ void gemm_in_bn(const uint16_t* __restrict__ xbf, const uint16_t* __restrict__ wbf,
                           const float* __restrict__ g, const float* __restrict__ bt,
                           const float* __restrict__ mn, const float* __restrict__ vr,
                           float* __restrict__ ident) {
  const int l  = threadIdx.x;
  const int n0 = blockIdx.x * 64, o0 = blockIdx.y * 16, b = blockIdx.z;
  const int lh = l & 15;
  const int kbB = (l < 16) ? 0 : 16;
  AccF acc[4];
#pragma unroll
  for (int t = 0; t < 4; ++t) acc[t].v = v8f_zero();
  const uint16_t* wr = wbf + (size_t)(o0 + lh) * DIMC;
  const uint16_t* xb = xbf + ((size_t)b * NTOK + n0 + lh) * DIMC;
  for (int c0 = 0; c0 < DIMC; c0 += 32) {
    ABf A;
#pragma unroll
    for (int v = 0; v < 8; ++v)
      A.u[v] = *(const uint32_t*)(wr + c0 + a_k0(v, l));
#pragma unroll
    for (int t = 0; t < 4; ++t) {
      ABf B;
      const uint16_t* xr = xb + (size_t)t * 16 * DIMC + c0 + kbB;
#pragma unroll
      for (int v = 0; v < 8; ++v) B.u[v] = *(const uint32_t*)(xr + 2 * v);
      acc[t].v = __builtin_amdgcn_wmma_f32_16x16x32_bf16(false, A.v, false, B.v,
                                                         (short)0, acc[t].v, false, false);
    }
  }
  const int hl = (l < 16) ? 0 : 8;
#pragma unroll
  for (int v = 0; v < 8; ++v) {
    int o = o0 + v + hl;
    float inv  = g[o] * rsqrtf(vr[o] + EPSF);
    float beta = bt[o] - mn[o] * inv;
    float* op = ident + ((size_t)b * DIMC + o) * NTOK + n0 + lh;
#pragma unroll
    for (int t = 0; t < 4; ++t) op[t * 16] = acc[t].f[v] * inv + beta;
  }
}

// ---------------- Kernel 2: depthwise 3x3, SAME, writes bf16 tT[b][n][c] -----------
__global__ void dwconv(const float* __restrict__ in, long bstride, int coff,
                       const float* __restrict__ wdw, uint16_t* __restrict__ tT) {
  int idx = blockIdx.x * blockDim.x + threadIdx.x;       // B*C*N threads
  int n = idx & (NTOK - 1);
  int c = (idx >> 10) & (CCH - 1);
  int b = idx >> 17;
  int h = n >> 5, w = n & 31;
  const float* ip = in + (size_t)b * bstride + (size_t)(coff + c) * NTOK;
  const float* wp = wdw + c * 9;
  float s = 0.f;
#pragma unroll
  for (int dy = 0; dy < 3; ++dy) {
    int ih = h + dy - 1;
    if (ih < 0 || ih >= LSZ) continue;
#pragma unroll
    for (int dx = 0; dx < 3; ++dx) {
      int iw = w + dx - 1;
      if (iw < 0 || iw >= LSZ) continue;
      s += wp[dy * 3 + dx] * ip[ih * LSZ + iw];
    }
  }
  tT[((size_t)b * NTOK + n) * CCH + c] = f2bf(s);
}

// ---------------- Kernel 3: pointwise GEMM + BN2 + residual for next stage ---------
// One wave: 16 (o) x 64 (n), A = w_pw tile shared.
__global__ void pw_bn(const uint16_t* __restrict__ tT, const uint16_t* __restrict__ wpw,
                      const float* __restrict__ g, const float* __restrict__ bt,
                      const float* __restrict__ mn, const float* __restrict__ vr,
                      const float* __restrict__ ident, int s,
                      float* __restrict__ y, float* __restrict__ nextin) {
  const int l  = threadIdx.x;
  const int n0 = blockIdx.x * 64, o0 = blockIdx.y * 16, b = blockIdx.z;
  const int lh = l & 15;
  const int kbB = (l < 16) ? 0 : 16;
  AccF acc[4];
#pragma unroll
  for (int t = 0; t < 4; ++t) acc[t].v = v8f_zero();
  const uint16_t* wr = wpw + (size_t)(o0 + lh) * CCH;
  const uint16_t* tb = tT + ((size_t)b * NTOK + n0 + lh) * CCH;
  for (int c0 = 0; c0 < CCH; c0 += 32) {
    ABf A;
#pragma unroll
    for (int v = 0; v < 8; ++v)
      A.u[v] = *(const uint32_t*)(wr + c0 + a_k0(v, l));
#pragma unroll
    for (int t = 0; t < 4; ++t) {
      ABf B;
      const uint16_t* tr = tb + (size_t)t * 16 * CCH + c0 + kbB;
#pragma unroll
      for (int v = 0; v < 8; ++v) B.u[v] = *(const uint32_t*)(tr + 2 * v);
      acc[t].v = __builtin_amdgcn_wmma_f32_16x16x32_bf16(false, A.v, false, B.v,
                                                         (short)0, acc[t].v, false, false);
    }
  }
  const int hl = (l < 16) ? 0 : 8;
#pragma unroll
  for (int v = 0; v < 8; ++v) {
    int o = o0 + v + hl;
    float inv  = g[o] * rsqrtf(vr[o] + EPSF);
    float beta = bt[o] - mn[o] * inv;
    float* yp = y + (((size_t)b * 4 + s) * CCH + o) * NTOK + n0 + lh;
    const float* idp = ident + ((size_t)b * DIMC + (s + 1) * CCH + o) * NTOK + n0 + lh;
    float* np = nextin + ((size_t)b * CCH + o) * NTOK + n0 + lh;
#pragma unroll
    for (int t = 0; t < 4; ++t) {
      float val = acc[t].f[v] * inv + beta;
      yp[t * 16] = val;
      if (s < 3) np[t * 16] = idp[t * 16] + val;
    }
  }
}

// ---------------- Kernel 4: pack Q [b,s,h,n,d] and V^T [b,s,h,d,n] as bf16 ----------
__global__ void pack_qv(const float* __restrict__ ident, const float* __restrict__ y,
                        uint16_t* __restrict__ Q, uint16_t* __restrict__ Vt) {
  int idx = blockIdx.x * blockDim.x + threadIdx.x;       // B*4*C*N threads
  int n = idx & (NTOK - 1);
  int c = (idx >> 10) & (CCH - 1);
  int s = (idx >> 17) & 3;
  int b = idx >> 19;
  float val = (s == 0) ? ident[((size_t)b * DIMC + c) * NTOK + n]
                       : y[(((size_t)b * 4 + s) * CCH + c) * NTOK + n];
  int h = c >> 5, d = c & 31;
  uint16_t bv = f2bf(val);
  size_t base = ((size_t)b * 4 + s) * NH + h;
  Q [(base * NTOK + n) * DH + d] = bv;
  Vt[(base * DH + d) * NTOK + n] = bv;
}

// ---------------- Kernel 5: flash self-attention (q=k=v), 16 rows / wave ------------
__global__ void attn(const uint16_t* __restrict__ Qm, const uint16_t* __restrict__ Vtm,
                     uint16_t* __restrict__ Out) {
  __shared__ __align__(16) uint16_t Pld[16 * 32];
  const int l   = threadIdx.x;
  const int blk = blockIdx.x;                  // ((b*4+s)*NH+h)*64 + qtile
  const int q0  = (blk & 63) * 16;
  const int h   = (blk >> 6) & 3;
  const int s   = (blk >> 8) & 3;
  const int b   = blk >> 10;
  const uint16_t* Q  = Qm  + (((size_t)b * 4 + s) * NH + h) * (size_t)NTOK * DH;
  const uint16_t* Vt = Vtm + (((size_t)b * 4 + s) * NH + h) * (size_t)DH * NTOK;
  const int lh = l & 15;
  const int kbB = (l < 16) ? 0 : 16;
  const int hl = (l < 16) ? 0 : 8;

  ABf aQ;
  const uint16_t* qr = Q + (size_t)(q0 + lh) * DH;
#pragma unroll
  for (int v = 0; v < 8; ++v)
    aQ.u[v] = *(const uint32_t*)(qr + a_k0(v, l));

  float rm[8], rl[8];
  AccF oa0, oa1; oa0.v = v8f_zero(); oa1.v = v8f_zero();
#pragma unroll
  for (int v = 0; v < 8; ++v) { rm[v] = -1e30f; rl[v] = 0.f; }

  for (int j0 = 0; j0 < NTOK; j0 += 32) {
    // ---- S = Q @ K^T (two 16-col tiles) ----
    ABf bK0, bK1;
    const uint16_t* kr0 = Q + (size_t)(j0 + lh) * DH + kbB;
    const uint16_t* kr1 = Q + (size_t)(j0 + 16 + lh) * DH + kbB;
#pragma unroll
    for (int v = 0; v < 8; ++v) {
      bK0.u[v] = *(const uint32_t*)(kr0 + 2 * v);
      bK1.u[v] = *(const uint32_t*)(kr1 + 2 * v);
    }
    AccF s0t, s1t;
    s0t.v = __builtin_amdgcn_wmma_f32_16x16x32_bf16(false, aQ.v, false, bK0.v,
                                                    (short)0, v8f_zero(), false, false);
    s1t.v = __builtin_amdgcn_wmma_f32_16x16x32_bf16(false, aQ.v, false, bK1.v,
                                                    (short)0, v8f_zero(), false, false);
    // ---- online softmax (row stats via 16-lane xor reductions) ----
    float p0[8], p1[8];
#pragma unroll
    for (int v = 0; v < 8; ++v) {
      float a0 = s0t.f[v] * ATT_SCALE, a1 = s1t.f[v] * ATT_SCALE;
      float mx = hmax16(fmaxf(a0, a1));
      float mnew = fmaxf(rm[v], mx);
      p0[v] = __expf(a0 - mnew);
      p1[v] = __expf(a1 - mnew);
      float sum = hsum16(p0[v] + p1[v]);
      float fac = __expf(rm[v] - mnew);
      rl[v] = rl[v] * fac + sum;
      rm[v] = mnew;
      oa0.f[v] *= fac;
      oa1.f[v] *= fac;
    }
    // ---- transpose P (D-layout -> A-layout) through LDS ----
#pragma unroll
    for (int v = 0; v < 8; ++v) {
      int m = v + hl;
      Pld[m * 32 + lh]      = f2bf(p0[v]);
      Pld[m * 32 + 16 + lh] = f2bf(p1[v]);
    }
    __syncthreads();
    ABf aP;
#pragma unroll
    for (int v = 0; v < 8; ++v)
      aP.u[v] = *(const uint32_t*)&Pld[lh * 32 + a_k0(v, l)];
    __syncthreads();
    // ---- O += P @ V (two d halves) ----
    ABf bV0, bV1;
    const uint16_t* vr0 = Vt + (size_t)lh * NTOK + j0 + kbB;
    const uint16_t* vr1 = Vt + (size_t)(16 + lh) * NTOK + j0 + kbB;
#pragma unroll
    for (int v = 0; v < 8; ++v) {
      bV0.u[v] = *(const uint32_t*)(vr0 + 2 * v);
      bV1.u[v] = *(const uint32_t*)(vr1 + 2 * v);
    }
    oa0.v = __builtin_amdgcn_wmma_f32_16x16x32_bf16(false, aP.v, false, bV0.v,
                                                    (short)0, oa0.v, false, false);
    oa1.v = __builtin_amdgcn_wmma_f32_16x16x32_bf16(false, aP.v, false, bV1.v,
                                                    (short)0, oa1.v, false, false);
  }
  // ---- normalize + scatter to [b, n, s*128 + h*32 + d] bf16 ----
#pragma unroll
  for (int v = 0; v < 8; ++v) {
    int nrow = q0 + v + hl;
    float inv = 1.0f / rl[v];
    size_t ob = ((size_t)b * NTOK + nrow) * DIMC + s * CCH + h * DH;
    Out[ob + lh]      = f2bf(oa0.f[v] * inv);
    Out[ob + 16 + lh] = f2bf(oa1.f[v] * inv);
  }
}

// ---------------- Kernel 6: out = AttOut @ W_out^T ---------------------------------
// One wave: 16 (n) x 64 (o), A = AttOut tile shared across 4 weight B-tiles.
__global__ void gemm_out(const uint16_t* __restrict__ A, const uint16_t* __restrict__ wbf,
                         float* __restrict__ out) {
  const int l  = threadIdx.x;
  const int o0 = blockIdx.x * 64, n0 = blockIdx.y * 16, b = blockIdx.z;
  const int lh = l & 15;
  const int kbB = (l < 16) ? 0 : 16;
  AccF acc[4];
#pragma unroll
  for (int t = 0; t < 4; ++t) acc[t].v = v8f_zero();
  const uint16_t* ar = A + ((size_t)b * NTOK + n0 + lh) * DIMC;
  const uint16_t* wb = wbf + (size_t)(o0 + lh) * DIMC;
  for (int c0 = 0; c0 < DIMC; c0 += 32) {
    ABf Am;
#pragma unroll
    for (int v = 0; v < 8; ++v)
      Am.u[v] = *(const uint32_t*)(ar + c0 + a_k0(v, l));
#pragma unroll
    for (int t = 0; t < 4; ++t) {
      ABf Bm;
      const uint16_t* wr = wb + (size_t)t * 16 * DIMC + c0 + kbB;
#pragma unroll
      for (int v = 0; v < 8; ++v) Bm.u[v] = *(const uint32_t*)(wr + 2 * v);
      acc[t].v = __builtin_amdgcn_wmma_f32_16x16x32_bf16(false, Am.v, false, Bm.v,
                                                         (short)0, acc[t].v, false, false);
    }
  }
  const int hl = (l < 16) ? 0 : 8;
#pragma unroll
  for (int v = 0; v < 8; ++v) {
    float* op = out + ((size_t)b * NTOK + n0 + v + hl) * DIMC + o0 + lh;
#pragma unroll
    for (int t = 0; t < 4; ++t) op[t * 16] = acc[t].f[v];
  }
}

extern "C" void kernel_launch(void* const* d_in, const int* in_sizes, int n_in,
                              void* d_out, int out_size, void* d_ws, size_t ws_size,
                              hipStream_t stream) {
  const float* x    = (const float*)d_in[0];
  const float* w_in = (const float*)d_in[1];
  const float* b1g  = (const float*)d_in[2];
  const float* b1b  = (const float*)d_in[3];
  const float* b1m  = (const float*)d_in[4];
  const float* b1v  = (const float*)d_in[5];
  const float* w_dw = (const float*)d_in[6];
  const float* w_pw = (const float*)d_in[7];
  const float* b2g  = (const float*)d_in[8];
  const float* b2b  = (const float*)d_in[9];
  const float* b2m  = (const float*)d_in[10];
  const float* b2v  = (const float*)d_in[11];
  const float* w_out= (const float*)d_in[12];

  char* ws = (char*)d_ws;
  size_t off = 0;
  auto take = [&](size_t bytes) { char* p = ws + off; off = (off + bytes + 255) & ~(size_t)255; return p; };
  float*    ident  = (float*)   take((size_t)BB * DIMC * NTOK * 4);        // 16 MB
  float*    ybuf   = (float*)   take((size_t)BB * 4 * CCH * NTOK * 4);     // 16 MB
  uint16_t* tT     = (uint16_t*)take((size_t)BB * NTOK * CCH * 2);         //  2 MB
  float*    convin = (float*)   take((size_t)BB * CCH * NTOK * 4);         //  4 MB
  uint16_t* Qm     = (uint16_t*)take((size_t)BB * 4 * NH * NTOK * DH * 2); //  8 MB
  uint16_t* Vtm    = (uint16_t*)take((size_t)BB * 4 * NH * DH * NTOK * 2); //  8 MB
  uint16_t* xbf    = (uint16_t*)take((size_t)BB * NTOK * DIMC * 2);        //  8 MB
  uint16_t* w_in_bf  = (uint16_t*)take((size_t)DIMC * DIMC * 2);           // 512 KB
  uint16_t* w_pw_bf  = (uint16_t*)take((size_t)CCH * CCH * 2);             //  32 KB
  uint16_t* w_out_bf = (uint16_t*)take((size_t)DIMC * DIMC * 2);           // 512 KB
  uint16_t* AttOut = (uint16_t*)ident; // ident is dead after pack_qv; reuse (8 MB needed)

  // 0) one-time bf16 conversions (L2-resident, trivial traffic)
  cvt_bf16<<<(BB * NTOK * DIMC) / 256, 256, 0, stream>>>(x, xbf, BB * NTOK * DIMC);
  cvt_bf16<<<(DIMC * DIMC) / 256, 256, 0, stream>>>(w_in, w_in_bf, DIMC * DIMC);
  cvt_bf16<<<(CCH * CCH) / 256, 256, 0, stream>>>(w_pw, w_pw_bf, CCH * CCH);
  cvt_bf16<<<(DIMC * DIMC) / 256, 256, 0, stream>>>(w_out, w_out_bf, DIMC * DIMC);

  // 1) input projection + BN1
  gemm_in_bn<<<dim3(NTOK / 64, DIMC / 16, BB), 32, 0, stream>>>(
      xbf, w_in_bf, b1g, b1b, b1m, b1v, ident);

  // 2) Res2Net stages s = 1..3 (sequential)
  for (int s = 1; s <= 3; ++s) {
    if (s == 1)
      dwconv<<<(BB * CCH * NTOK) / 256, 256, 0, stream>>>(
          ident, (long)DIMC * NTOK, CCH /*channels 128..255*/, w_dw, tT);
    else
      dwconv<<<(BB * CCH * NTOK) / 256, 256, 0, stream>>>(
          convin, (long)CCH * NTOK, 0, w_dw, tT);
    pw_bn<<<dim3(NTOK / 64, CCH / 16, BB), 32, 0, stream>>>(
        tT, w_pw_bf, b2g, b2b, b2m, b2v, ident, s, ybuf, convin);
  }

  // 3) pack bf16 Q and V^T
  pack_qv<<<(BB * 4 * CCH * NTOK) / 256, 256, 0, stream>>>(ident, ybuf, Qm, Vtm);

  // 4) attention (128 problems x 64 query tiles)
  attn<<<BB * 4 * NH * (NTOK / 16), 32, 0, stream>>>(Qm, Vtm, AttOut);

  // 5) output projection
  gemm_out<<<dim3(DIMC / 64, NTOK / 16, BB), 32, 0, stream>>>(
      AttOut, w_out_bf, (float*)d_out);
}
